// CausalSelfAttention_54030688584063
// MI455X (gfx1250) — matrix operand order repaired and checked by reading, not compiled, not run on hardware
//
#include <hip/hip_runtime.h>

typedef __attribute__((ext_vector_type(16))) __bf16 v16bf;
typedef __attribute__((ext_vector_type(8)))  float  v8f;
typedef __attribute__((ext_vector_type(4)))  unsigned int uint4v;
typedef __attribute__((ext_vector_type(8)))  int int8v;
typedef __attribute__((ext_vector_type(4)))  int int4v;

union AFrag { v16bf v; unsigned int u[8]; };
union CFrag { v8f v; float f[8]; };

__device__ __forceinline__ unsigned short f32_to_bf16(float x) {
    unsigned int bits = __float_as_uint(x);
    bits += 0x7FFFu + ((bits >> 16) & 1u);   // round-to-nearest-even
    return (unsigned short)(bits >> 16);
}

// ------------------------------------------------------------- TDM issue
// Pack D# groups per CDNA5 ISA ch.8 and issue TENSOR_LOAD_TO_LDS.
// 2D tile: tile0 elems/row, tile1 rows, row stride stride0 (elements).
// 1D copy: tile1 = 0.
__device__ __forceinline__ void tdm_load_to_lds(
    const void* gptr, unsigned lds_byte_off,
    unsigned tile0, unsigned tile1,
    unsigned long long stride0_elems,
    unsigned tensor0, unsigned tensor1)
{
    unsigned long long ga = (unsigned long long)(size_t)gptr;
    uint4v g0;
    g0.x = 1u;                                          // count=1, user D#
    g0.y = lds_byte_off;                                // lds_addr
    g0.z = (unsigned)(ga & 0xFFFFFFFFull);              // global_addr[31:0]
    g0.w = (unsigned)((ga >> 32) & 0x1FFFFFFull)        // global_addr[56:32]
         | (2u << 30);                                  // type = 2 (image)
    int8v g1;
    g1[0] = (int)(1u << 16);                            // data_size = 2 bytes
    g1[1] = (int)((tensor0 & 0xFFFFu) << 16);           // tensor_dim0[15:0]
    g1[2] = (int)(((tensor0 >> 16) & 0xFFFFu)
          | ((tensor1 & 0xFFFFu) << 16));
    g1[3] = (int)(((tensor1 >> 16) & 0xFFFFu)
          | ((tile0 & 0xFFFFu) << 16));                 // tile_dim0
    g1[4] = (int)(tile1 & 0xFFFFu);                     // tile_dim1, tile_dim2=0
    g1[5] = (int)(unsigned)(stride0_elems & 0xFFFFFFFFull);
    g1[6] = (int)(unsigned)((stride0_elems >> 32) & 0xFFFFull); // stride1 = 0
    g1[7] = 0;
    int4v z4 = {0, 0, 0, 0};
    int8v z8 = {0, 0, 0, 0, 0, 0, 0, 0};
    __builtin_amdgcn_tensor_load_to_lds(g0, g1, z4, z4, z8, 0);
}

// ---------------------------------------------------------------- convert
__global__ void cvt_bf16_kernel(const float* __restrict__ in,
                                unsigned short* __restrict__ out, int n) {
    int i = blockIdx.x * blockDim.x + threadIdx.x;
    int stride = gridDim.x * blockDim.x;
    for (; i < n; i += stride) out[i] = f32_to_bf16(in[i]);
}

// ---------------------------------------------------------------- GEMM
// C[M,N] = A[M,K] @ W[N,K]^T (bf16, K-contiguous). One wave -> 32x64 tile:
// every B fragment reused by two WMMAs. mode 0: bf16 row-major; mode 1: Q/K
// scatter [B,H,T,D]; mode 2: V scatter [B,H,D,T]; mode 3: fp32 row-major.
__global__ __launch_bounds__(256) void gemm_bf16_kernel(
    const unsigned short* __restrict__ A,
    const unsigned short* __restrict__ W,
    unsigned short* __restrict__ outb, float* __restrict__ outf,
    int M, int N, int K, int mode)
{
    const int lane = threadIdx.x & 31;
    const int wave = threadIdx.x >> 5;
    const int task = blockIdx.x * 8 + wave;
    const int nt64 = N >> 6;
    const int mtile = task / nt64;
    const int ntile = task % nt64;
    const int m0 = mtile * 32;
    if (m0 >= M) return;
    const int half = lane >> 4;
    const int l15  = lane & 15;

    const unsigned int* Arow0 =
        (const unsigned int*)(A + (size_t)(m0 + l15) * K);
    const unsigned int* Arow1 =
        (const unsigned int*)(A + (size_t)(m0 + 16 + l15) * K);

    CFrag acc[2][4];
#pragma unroll
    for (int i2 = 0; i2 < 2; ++i2)
#pragma unroll
        for (int j = 0; j < 4; ++j)
#pragma unroll
            for (int r = 0; r < 8; ++r) acc[i2][j].f[r] = 0.f;

    for (int kk = 0; kk < K; kk += 32) {
        AFrag a0, a1;
#pragma unroll
        for (int v2 = 0; v2 < 8; ++v2) {
            int k = kk + ((v2 < 4) ? (8 * half + 2 * v2)
                                   : (16 + 8 * half + 2 * (v2 - 4)));
            a0.u[v2] = Arow0[k >> 1];
            a1.u[v2] = Arow1[k >> 1];
        }
#pragma unroll
        for (int j = 0; j < 4; ++j) {
            const unsigned int* Wrow = (const unsigned int*)
                (W + (size_t)(ntile * 64 + j * 16 + l15) * K);
            AFrag b;
#pragma unroll
            for (int v2 = 0; v2 < 8; ++v2) {
                int k = kk + ((v2 < 4) ? (8 * half + 2 * v2)
                                       : (16 + 8 * half + 2 * (v2 - 4)));
                b.u[v2] = Wrow[k >> 1];
            }
            acc[0][j].v = __builtin_amdgcn_wmma_f32_16x16x32_bf16(
                false, a0.v, false, b.v, (short)0, acc[0][j].v, false, false);
            acc[1][j].v = __builtin_amdgcn_wmma_f32_16x16x32_bf16(
                false, a1.v, false, b.v, (short)0, acc[1][j].v, false, false);
        }
    }

#pragma unroll
    for (int i2 = 0; i2 < 2; ++i2)
#pragma unroll
        for (int j = 0; j < 4; ++j)
#pragma unroll
            for (int r = 0; r < 8; ++r) {
                int row = m0 + i2 * 16 + r + 8 * half;
                int col = ntile * 64 + j * 16 + l15;
                float val = acc[i2][j].f[r];
                if (mode == 3) {
                    outf[(size_t)row * N + col] = val;
                } else if (mode == 0) {
                    outb[(size_t)row * N + col] = f32_to_bf16(val);
                } else {
                    int b = row >> 11, t = row & 2047;   // T = 2048
                    int h = col >> 6,  d = col & 63;     // D = 64
                    size_t idx = (mode == 1)
                        ? ((size_t)((b * 16 + h) * 2048 + t)) * 64 + d
                        : ((size_t)((b * 16 + h) * 64 + d)) * 2048 + t;
                    outb[idx] = f32_to_bf16(val);
                }
            }
}

// ---------------------------------------------------------------- attention
// Q,K: [BH, T, 64] bf16; Vt: [BH, 64, T] bf16; O: [B, T, C] bf16.
// One wave per 16-row Q tile. K/V s-blocks (32 wide) streamed into per-wave
// double-buffered LDS by the Tensor Data Mover, overlapped with compute.
__global__ __launch_bounds__(256) void attn_kernel(
    const unsigned short* __restrict__ Q,
    const unsigned short* __restrict__ Kt,
    const unsigned short* __restrict__ Vt,
    unsigned short* __restrict__ O)
{
    // per wave: K dbl buf 2*2048, V dbl buf 2*2048, P tile 512 (ushorts)
    __shared__ unsigned short smem[8 * 8704];
    const int lane = threadIdx.x & 31;
    const int wave = threadIdx.x >> 5;
    const int task = blockIdx.x * 8 + wave;
    const int bh = task >> 7;          // 128 t-tiles per (b,h)
    const int t0 = (task & 127) * 16;
    const int half = lane >> 4;
    const int l15  = lane & 15;

    unsigned short* wbase = &smem[wave * 8704];
    unsigned short* kbuf  = wbase;            // [2][32][64]
    unsigned short* vbuf  = wbase + 4096;     // [2][64][32]
    unsigned short* pbuf  = wbase + 8192;     // [16][32]

    const unsigned short* qb = Q  + (size_t)bh * 2048 * 64;
    const unsigned short* kb = Kt + (size_t)bh * 2048 * 64;
    const unsigned short* vb = Vt + (size_t)bh * 64 * 2048;

    AFrag aq[2];
    {
        const unsigned int* qrow =
            (const unsigned int*)(qb + (size_t)(t0 + l15) * 64);
#pragma unroll
        for (int kk2 = 0; kk2 < 2; ++kk2)
#pragma unroll
            for (int v2 = 0; v2 < 8; ++v2) {
                int d = kk2 * 32 + ((v2 < 4) ? (8 * half + 2 * v2)
                                             : (16 + 8 * half + 2 * (v2 - 4)));
                aq[kk2].u[v2] = qrow[d >> 1];
            }
    }

    CFrag o[4];
#pragma unroll
    for (int j = 0; j < 4; ++j)
#pragma unroll
        for (int r = 0; r < 8; ++r) o[j].f[r] = 0.f;
    float mrow[8], lrow[8];
#pragma unroll
    for (int r = 0; r < 8; ++r) { mrow[r] = -3.0e38f; lrow[r] = 0.f; }

    const float scale = 0.125f;   // 1/sqrt(64)
    const int nblocks = (t0 + 15) / 32 + 1;

    // prologue: TDM-load s-block 0 into buffer 0
    tdm_load_to_lds(kb, (unsigned)(size_t)(void*)kbuf,
                    2048, 0, 2048ull, 2048, 1);            // 4KB linear
    tdm_load_to_lds(vb, (unsigned)(size_t)(void*)vbuf,
                    32, 64, 2048ull, 2048, 64);            // 64 rows x 32

    for (int i = 0; i < nblocks; ++i) {
        const int s0  = i * 32;
        const int buf = i & 1;
        unsigned short* kcur = kbuf + buf * 2048;
        unsigned short* vcur = vbuf + buf * 2048;

        if (i + 1 < nblocks) {
            // protect previous fragment reads of the buffer being refilled
            asm volatile("s_wait_dscnt 0" ::: "memory");
            unsigned short* knxt = kbuf + (buf ^ 1) * 2048;
            unsigned short* vnxt = vbuf + (buf ^ 1) * 2048;
            tdm_load_to_lds(kb + (size_t)(s0 + 32) * 64,
                            (unsigned)(size_t)(void*)knxt,
                            2048, 0, 2048ull, 2048, 1);
            tdm_load_to_lds(vb + (s0 + 32),
                            (unsigned)(size_t)(void*)vnxt,
                            32, 64, 2048ull, 2048, 64);
            __builtin_amdgcn_s_wait_tensorcnt(2);  // current pair complete
        } else {
            __builtin_amdgcn_s_wait_tensorcnt(0);
        }
        asm volatile("" ::: "memory");

        // scores: S = Q @ K^T from LDS K tile [32][64]
        CFrag sc[2];
#pragma unroll
        for (int st = 0; st < 2; ++st) {
#pragma unroll
            for (int r = 0; r < 8; ++r) sc[st].f[r] = 0.f;
            const unsigned int* krow =
                (const unsigned int*)(kcur + (st * 16 + l15) * 64);
#pragma unroll
            for (int kk2 = 0; kk2 < 2; ++kk2) {
                AFrag bk;
#pragma unroll
                for (int v2 = 0; v2 < 8; ++v2) {
                    int d = kk2 * 32 + ((v2 < 4) ? (8 * half + 2 * v2)
                                                 : (16 + 8 * half + 2 * (v2 - 4)));
                    bk.u[v2] = krow[d >> 1];
                }
                sc[st].v = __builtin_amdgcn_wmma_f32_16x16x32_bf16(
                    false, aq[kk2].v, false, bk.v, (short)0, sc[st].v,
                    false, false);
            }
        }

        // scale + causal mask
        float p[2][8];
#pragma unroll
        for (int st = 0; st < 2; ++st) {
            int s = s0 + st * 16 + l15;
#pragma unroll
            for (int r = 0; r < 8; ++r) {
                int t = t0 + r + 8 * half;
                float v = sc[st].f[r] * scale;
                p[st][r] = (s <= t) ? v : -3.0e38f;
            }
        }

        // online softmax (row lives in one 16-lane half)
#pragma unroll
        for (int r = 0; r < 8; ++r) {
            float rm = fmaxf(p[0][r], p[1][r]);
#pragma unroll
            for (int off = 8; off >= 1; off >>= 1)
                rm = fmaxf(rm, __shfl_xor(rm, off, 32));
            float mnew  = fmaxf(mrow[r], rm);
            float alpha = __expf(mrow[r] - mnew);
            float e0 = __expf(p[0][r] - mnew);
            float e1 = __expf(p[1][r] - mnew);
            p[0][r] = e0; p[1][r] = e1;
            float rs = e0 + e1;
#pragma unroll
            for (int off = 8; off >= 1; off >>= 1)
                rs += __shfl_xor(rs, off, 32);
            lrow[r] = lrow[r] * alpha + rs;
            mrow[r] = mnew;
#pragma unroll
            for (int j = 0; j < 4; ++j) o[j].f[r] *= alpha;
        }

        // P (C-layout f32) -> LDS bf16 -> A-fragment re-layout (same wave)
#pragma unroll
        for (int st = 0; st < 2; ++st)
#pragma unroll
            for (int r = 0; r < 8; ++r)
                pbuf[(r + 8 * half) * 32 + st * 16 + l15] =
                    f32_to_bf16(p[st][r]);
        asm volatile("s_wait_dscnt 0" ::: "memory");

        AFrag ap;
        {
            const unsigned int* prow = (const unsigned int*)(pbuf + l15 * 32);
#pragma unroll
            for (int v2 = 0; v2 < 8; ++v2) {
                int k = (v2 < 4) ? (8 * half + 2 * v2)
                                 : (16 + 8 * half + 2 * (v2 - 4));
                ap.u[v2] = prow[k >> 1];
            }
        }

        // O += P @ V from LDS V tile [64][32]
#pragma unroll
        for (int j = 0; j < 4; ++j) {
            const unsigned int* vrow =
                (const unsigned int*)(vcur + (j * 16 + l15) * 32);
            AFrag bv;
#pragma unroll
            for (int v2 = 0; v2 < 8; ++v2) {
                int sk = (v2 < 4) ? (8 * half + 2 * v2)
                                  : (16 + 8 * half + 2 * (v2 - 4));
                bv.u[v2] = vrow[sk >> 1];
            }
            o[j].v = __builtin_amdgcn_wmma_f32_16x16x32_bf16(
                false, ap.v, false, bv.v, (short)0, o[j].v, false, false);
        }
    }

    // normalize and scatter to [B, T, C] bf16
    int b = bh >> 4, h = bh & 15;
#pragma unroll
    for (int r = 0; r < 8; ++r) {
        float inv = 1.0f / lrow[r];
        int t = t0 + r + 8 * half;
#pragma unroll
        for (int j = 0; j < 4; ++j) {
            int c = h * 64 + j * 16 + l15;
            O[((size_t)(b * 2048 + t)) * 1024 + c] =
                f32_to_bf16(o[j].f[r] * inv);
        }
    }
}

// ---------------------------------------------------------------- launch
extern "C" void kernel_launch(void* const* d_in, const int* in_sizes, int n_in,
                              void* d_out, int out_size, void* d_ws, size_t ws_size,
                              hipStream_t stream) {
    const float* x  = (const float*)d_in[0];
    const float* wq = (const float*)d_in[1];
    const float* wk = (const float*)d_in[2];
    const float* wv = (const float*)d_in[3];
    const float* wo = (const float*)d_in[4];

    char* ws = (char*)d_ws;
    unsigned short* xb  = (unsigned short*)(ws + 0);          // 16 MB
    unsigned short* wqb = (unsigned short*)(ws + 16777216);   //  2 MB each
    unsigned short* wkb = (unsigned short*)(ws + 18874368);
    unsigned short* wvb = (unsigned short*)(ws + 20971520);
    unsigned short* wob = (unsigned short*)(ws + 23068672);
    unsigned short* qB  = (unsigned short*)(ws + 25165824);   // 16 MB
    unsigned short* kB  = (unsigned short*)(ws + 41943040);   // 16 MB
    unsigned short* vB  = (unsigned short*)(ws + 58720256);   // 16 MB
    unsigned short* attn = xb;                                // reuse

    const int M = 4 * 2048, C = 1024;

    cvt_bf16_kernel<<<2048, 256, 0, stream>>>(x,  xb,  M * C);
    cvt_bf16_kernel<<<1024, 256, 0, stream>>>(wq, wqb, C * C);
    cvt_bf16_kernel<<<1024, 256, 0, stream>>>(wk, wkb, C * C);
    cvt_bf16_kernel<<<1024, 256, 0, stream>>>(wv, wvb, C * C);
    cvt_bf16_kernel<<<1024, 256, 0, stream>>>(wo, wob, C * C);

    // QKV projections: waves = (M/32)*(C/64) = 4096 -> 512 blocks
    gemm_bf16_kernel<<<512, 256, 0, stream>>>(xb, wqb, qB, nullptr, M, C, C, 1);
    gemm_bf16_kernel<<<512, 256, 0, stream>>>(xb, wkb, kB, nullptr, M, C, C, 1);
    gemm_bf16_kernel<<<512, 256, 0, stream>>>(xb, wvb, vB, nullptr, M, C, C, 2);

    // attention: 64 (b,h) * 128 t-tiles = 8192 wave tasks -> 1024 blocks
    attn_kernel<<<1024, 256, 0, stream>>>(qB, kB, vB, attn);

    // output projection -> fp32 d_out
    gemm_bf16_kernel<<<512, 256, 0, stream>>>(attn, wob, nullptr, (float*)d_out,
                                              M, C, C, 3);
}
